// MultiHeadAttention_V2_51616916964141
// MI455X (gfx1250) — matrix operand order repaired and checked by reading, not compiled, not run on hardware
//
#include <hip/hip_runtime.h>
#include <stdint.h>

#define B_  2
#define S_  2048
#define D_  1024
#define H_  16
#define HD_ 64

typedef __attribute__((ext_vector_type(16))) __bf16 bf16x16;
typedef __attribute__((ext_vector_type(8)))  float  f32x8;

union FragAB { bf16x16 v; uint4 q[2]; };
union FragC  { f32x8   v; float f[8]; };

__device__ __forceinline__ uint16_t f2bf(float f) {
  uint32_t u = __float_as_uint(f);
  u += 0x7FFFu + ((u >> 16) & 1u);   // round-to-nearest-even
  return (uint16_t)(u >> 16);
}

// ---------------- async LDS staging (ASYNCcnt path), with sync fallback ----
typedef int v4i __attribute__((__vector_size__(16)));
typedef __attribute__((address_space(1))) v4i* g_v4i_ptr;   // global int4*
typedef __attribute__((address_space(3))) v4i* l_v4i_ptr;   // LDS int4*

#if __has_builtin(__builtin_amdgcn_global_load_async_to_lds_b128) && \
    __has_builtin(__builtin_amdgcn_s_wait_asynccnt)
#define USE_ASYNC_LDS 1
#endif

__device__ __forceinline__ void cp16(void* l, const void* g) {
#ifdef USE_ASYNC_LDS
  __builtin_amdgcn_global_load_async_to_lds_b128(
      (g_v4i_ptr)(uintptr_t)g, (l_v4i_ptr)(uint32_t)(uintptr_t)l, 0, 0);
#else
  *(uint4*)l = *(const uint4*)g;
#endif
}
__device__ __forceinline__ void stage_wait() {
#ifdef USE_ASYNC_LDS
  __builtin_amdgcn_s_wait_asynccnt(0);
#endif
  __syncthreads();
}

// ---------------- DPP16 cross-lane reductions (rows of 16 lanes) ----------
template<int CTRL>
__device__ __forceinline__ float dpp_movf(float x) {
  return __uint_as_float((unsigned)__builtin_amdgcn_update_dpp(
      0, (int)__float_as_uint(x), CTRL, 0xF, 0xF, true));
}
__device__ __forceinline__ float red_max16(float x) {
  x = fmaxf(x, dpp_movf<0xB1>(x));    // quad_perm xor1
  x = fmaxf(x, dpp_movf<0x4E>(x));    // quad_perm xor2
  x = fmaxf(x, dpp_movf<0x141>(x));   // row_half_mirror
  x = fmaxf(x, dpp_movf<0x140>(x));   // row_mirror
  return x;
}
__device__ __forceinline__ float red_sum16(float x) {
  x += dpp_movf<0xB1>(x);
  x += dpp_movf<0x4E>(x);
  x += dpp_movf<0x141>(x);
  x += dpp_movf<0x140>(x);
  return x;
}

// ---------------------------------------------------------------- converts
__global__ void cvt_f32_bf16(const float* __restrict__ src,
                             uint16_t* __restrict__ dst, int n) {
  int i = blockIdx.x * blockDim.x + threadIdx.x;
  if (i < n) dst[i] = f2bf(src[i]);
}

// W [K][N] f32  ->  Wt [N][K] bf16 (LDS tile transpose, one-time cost)
__global__ __launch_bounds__(256)
void cvt_t_f32_bf16(const float* __restrict__ W, uint16_t* __restrict__ Wt,
                    int K, int N) {
  __shared__ float tile[32][33];
  int tx = threadIdx.x, ty = threadIdx.y;          // (32, 8)
  int n0 = blockIdx.x * 32, k0 = blockIdx.y * 32;
#pragma unroll
  for (int j = 0; j < 4; j++)
    tile[ty + j * 8][tx] = W[(size_t)(k0 + ty + j * 8) * N + n0 + tx];
  __syncthreads();
#pragma unroll
  for (int j = 0; j < 4; j++)
    Wt[(size_t)(n0 + ty + j * 8) * K + k0 + tx] = f2bf(tile[tx][ty + j * 8]);
}

// ---------------------------------------------------------------- GEMM
// C[M,N] = A[M,K] * W[K,N] with W given pre-transposed (Wt[N][K], bf16).
// Block tile 128x128, 8 waves, wave tile 32x64, K-step 64 (16 WMMA/step/wave).
// MODE: 0 = bf16 row-major out, 1 = bf16 out written as Vt[(b,h,d)][s],
//       2 = f32 out + bias.
template<int MODE>
__global__ __launch_bounds__(256)
void gemm_bf16(const uint16_t* __restrict__ A,
               const uint16_t* __restrict__ Bw,   // Wt [N][K]
               const float* __restrict__ bias,
               void* __restrict__ Out,
               int M, int N, int K)
{
  __shared__ __align__(16) uint16_t As[128 * 64];  // [row][k]
  __shared__ __align__(16) uint16_t Bs[128 * 64];  // [n][k]

  const int tid  = threadIdx.x;
  const int wave = tid >> 5;
  const int lane = tid & 31;
  const int ln   = lane & 15;
  const int hi   = lane >> 4;

  const int m0 = blockIdx.y * 128;
  const int n0 = blockIdx.x * 128;
  const int wm = (wave & 3) * 32;
  const int wn = (wave >> 2) * 64;

  FragC acc[2][4];
#pragma unroll
  for (int i = 0; i < 2; i++)
#pragma unroll
    for (int j = 0; j < 4; j++)
#pragma unroll
      for (int r = 0; r < 8; r++) acc[i][j].f[r] = 0.0f;

  const int sr = tid >> 1;            // 0..127
  const int sc = (tid & 1) * 32;      // 0 / 32
  const uint16_t* srcA = A  + (size_t)(m0 + sr) * K + sc;
  const uint16_t* srcB = Bw + (size_t)(n0 + sr) * K + sc;

  for (int k0 = 0; k0 < K; k0 += 64) {
#pragma unroll
    for (int j = 0; j < 4; j++) {
      cp16(&As[sr * 64 + sc + j * 8], srcA + k0 + j * 8);
      cp16(&Bs[sr * 64 + sc + j * 8], srcB + k0 + j * 8);
    }
    stage_wait();
    if (k0 + 64 < K) {                 // prefetch next tiles
      __builtin_prefetch(srcA + k0 + 64, 0, 0);
      __builtin_prefetch(srcB + k0 + 64, 0, 0);
    }

#pragma unroll
    for (int kc = 0; kc < 64; kc += 32) {
      FragAB a[2], b[4];
#pragma unroll
      for (int i = 0; i < 2; i++) {
        int row = wm + i * 16 + ln;
        a[i].q[0] = *(const uint4*)&As[row * 64 + kc + hi * 8];
        a[i].q[1] = *(const uint4*)&As[row * 64 + kc + hi * 8 + 16];
      }
#pragma unroll
      for (int j = 0; j < 4; j++) {
        int col = wn + j * 16 + ln;
        b[j].q[0] = *(const uint4*)&Bs[col * 64 + kc + hi * 8];
        b[j].q[1] = *(const uint4*)&Bs[col * 64 + kc + hi * 8 + 16];
      }
#pragma unroll
      for (int i = 0; i < 2; i++)
#pragma unroll
        for (int j = 0; j < 4; j++)
          acc[i][j].v = __builtin_amdgcn_wmma_f32_16x16x32_bf16(
              false, a[i].v, false, b[j].v, (short)0, acc[i][j].v, false, false);
    }
    __syncthreads();
  }

#pragma unroll
  for (int i = 0; i < 2; i++)
#pragma unroll
    for (int j = 0; j < 4; j++) {
      int col = n0 + wn + j * 16 + ln;
      float bv = (MODE == 2) ? bias[col] : 0.0f;
#pragma unroll
      for (int r = 0; r < 8; r++) {
        int row = m0 + wm + i * 16 + r + hi * 8;
        float val = acc[i][j].f[r] + bv;
        if (MODE == 2) {
          ((float*)Out)[(size_t)row * N + col] = val;
        } else if (MODE == 1) {      // V -> Vt[((b*H+h)*64+d)][s]
          int b = row >> 11, s = row & (S_ - 1);
          int h = col >> 6,  d = col & (HD_ - 1);
          ((uint16_t*)Out)[((size_t)((b * H_ + h) * HD_ + d)) * S_ + s] = f2bf(val);
        } else {
          ((uint16_t*)Out)[(size_t)row * N + col] = f2bf(val);
        }
      }
    }
}

// ---------------------------------------------------------------- flash attention
// One WG per (b, h, 64-query tile). 4 waves x 16 query rows. Key blocks of 32.
__global__ __launch_bounds__(128)
void attn_kernel(const uint16_t* __restrict__ Q,
                 const uint16_t* __restrict__ Kb,
                 const uint16_t* __restrict__ VtG,  // [(b*H+h)*64+d][S]
                 uint16_t* __restrict__ Z)
{
  __shared__ __align__(16) uint16_t Ks[32 * 64];      // [key][d]
  __shared__ __align__(16) uint16_t Vt[64 * 32];      // [d][key]
  __shared__ __align__(16) uint16_t Ps[4][16 * 32];   // per-wave P staging

  const int tid  = threadIdx.x;
  const int wv   = tid >> 5;
  const int lane = tid & 31;
  const int ln   = lane & 15;
  const int hi   = lane >> 4;

  const int qt = blockIdx.x;
  const int h  = blockIdx.y;
  const int b  = blockIdx.z;

  const int q0    = qt * 64;
  const int qbase = q0 + wv * 16;
  const size_t rowQ = ((size_t)b * S_ + qbase) * D_ + h * HD_;
  const size_t vhead = ((size_t)(b * H_ + h)) * HD_ * S_;
  const float CEXP = 0.18033688011112042f;   // (1/8) * log2(e)

  FragAB qf[2];
#pragma unroll
  for (int t = 0; t < 2; t++) {
    const uint16_t* src = Q + rowQ + (size_t)ln * D_ + t * 32 + hi * 8;
    qf[t].q[0] = *(const uint4*)(src);
    qf[t].q[1] = *(const uint4*)(src + 16);
  }

  float m[8], l[8];
  FragC o[4];
#pragma unroll
  for (int r = 0; r < 8; r++) { m[r] = -1e30f; l[r] = 0.0f; }
#pragma unroll
  for (int t = 0; t < 4; t++)
#pragma unroll
    for (int r = 0; r < 8; r++) o[t].f[r] = 0.0f;

  const int qhi  = qbase + 15;
  const int kend = q0 + 64;

  for (int kb = 0; kb < kend; kb += 32) {
    // cooperative stage: waves 0-1 copy Vt rows, waves 2-3 copy K rows (b128)
    if (tid < 64) {
      const uint16_t* src = VtG + vhead + (size_t)tid * S_ + kb;
#pragma unroll
      for (int j = 0; j < 4; j++) cp16(&Vt[tid * 32 + j * 8], src + j * 8);
    } else {
      int u = tid - 64, key = u >> 1, c = (u & 1) * 32;
      const uint16_t* src = Kb + ((size_t)b * S_ + kb + key) * D_ + h * HD_ + c;
#pragma unroll
      for (int j = 0; j < 4; j++) cp16(&Ks[key * 64 + c + j * 8], src + j * 8);
    }
    stage_wait();

    if (kb <= qhi) {
      // scores: two 16(q)x16(key) tiles over d=64
      FragC s[2];
#pragma unroll
      for (int t = 0; t < 2; t++) {
        int key = t * 16 + ln;
        FragAB bk0, bk1;
        bk0.q[0] = *(const uint4*)&Ks[key * 64 + hi * 8];
        bk0.q[1] = *(const uint4*)&Ks[key * 64 + hi * 8 + 16];
        bk1.q[0] = *(const uint4*)&Ks[key * 64 + 32 + hi * 8];
        bk1.q[1] = *(const uint4*)&Ks[key * 64 + 32 + hi * 8 + 16];
        FragC z;
#pragma unroll
        for (int r = 0; r < 8; r++) z.f[r] = 0.0f;
        s[t].v = __builtin_amdgcn_wmma_f32_16x16x32_bf16(
            false, qf[0].v, false, bk0.v, (short)0, z.v, false, false);
        s[t].v = __builtin_amdgcn_wmma_f32_16x16x32_bf16(
            false, qf[1].v, false, bk1.v, (short)0, s[t].v, false, false);
      }

      // causal mask only on diagonal blocks (wave-uniform branch)
      if (kb + 31 > qbase) {
#pragma unroll
        for (int r = 0; r < 8; r++) {
          int qrow = qbase + r + hi * 8;
          s[0].f[r] = (kb + ln      <= qrow) ? s[0].f[r] : -1e30f;
          s[1].f[r] = (kb + 16 + ln <= qrow) ? s[1].f[r] : -1e30f;
        }
      }

      // online softmax in raw-score domain; exp2 with folded 0.125*log2(e)
#pragma unroll
      for (int r = 0; r < 8; r++) {
        float v0 = s[0].f[r], v1 = s[1].f[r];
        float mx   = red_max16(fmaxf(v0, v1));
        float mnew = fmaxf(m[r], mx);
        float corr = exp2f((m[r] - mnew) * CEXP);
        m[r] = mnew;
        l[r] *= corr;
#pragma unroll
        for (int t = 0; t < 4; t++) o[t].f[r] *= corr;
        float p0 = exp2f((v0 - mnew) * CEXP);
        float p1 = exp2f((v1 - mnew) * CEXP);
        l[r] += red_sum16(p0 + p1);
        int row = r + hi * 8;                  // C-layout -> LDS (bf16)
        Ps[wv][row * 32 + ln]      = f2bf(p0);
        Ps[wv][row * 32 + 16 + ln] = f2bf(p1);
      }

      // reload P as A-fragment (same-wave LDS is in-order)
      FragAB pf;
      pf.q[0] = *(const uint4*)&Ps[wv][ln * 32 + hi * 8];
      pf.q[1] = *(const uint4*)&Ps[wv][ln * 32 + hi * 8 + 16];

      // O += P * V
#pragma unroll
      for (int t = 0; t < 4; t++) {
        FragAB vf;
        vf.q[0] = *(const uint4*)&Vt[(t * 16 + ln) * 32 + hi * 8];
        vf.q[1] = *(const uint4*)&Vt[(t * 16 + ln) * 32 + hi * 8 + 16];
        o[t].v = __builtin_amdgcn_wmma_f32_16x16x32_bf16(
            false, pf.v, false, vf.v, (short)0, o[t].v, false, false);
      }
    }
    __syncthreads();
  }

#pragma unroll
  for (int r = 0; r < 8; r++) {
    float inv = 1.0f / l[r];
    int q = qbase + r + hi * 8;
    size_t base = ((size_t)b * S_ + q) * D_ + h * HD_;
#pragma unroll
    for (int t = 0; t < 4; t++)
      Z[base + t * 16 + ln] = f2bf(o[t].f[r] * inv);
  }
}

// ---------------------------------------------------------------- launcher
extern "C" void kernel_launch(void* const* d_in, const int* in_sizes, int n_in,
                              void* d_out, int out_size, void* d_ws, size_t ws_size,
                              hipStream_t stream) {
  const float* x  = (const float*)d_in[0];
  const float* Wq = (const float*)d_in[1];
  const float* Wk = (const float*)d_in[2];
  const float* Wv = (const float*)d_in[3];
  const float* Wo = (const float*)d_in[4];
  const float* bo = (const float*)d_in[5];
  float* out = (float*)d_out;

  const int M = B_ * S_;                 // 4096
  const size_t nx = (size_t)M * D_;
  const size_t nw = (size_t)D_ * D_;

  uint8_t* ws = (uint8_t*)d_ws;
  size_t off = 0;
  uint16_t* xb   = (uint16_t*)(ws + off); off += nx * 2;
  uint16_t* Wqt  = (uint16_t*)(ws + off); off += nw * 2;
  uint16_t* Wkt  = (uint16_t*)(ws + off); off += nw * 2;
  uint16_t* Wvt  = (uint16_t*)(ws + off); off += nw * 2;
  uint16_t* Wot  = (uint16_t*)(ws + off); off += nw * 2;
  uint16_t* Qb   = (uint16_t*)(ws + off); off += nx * 2;
  uint16_t* Kb   = (uint16_t*)(ws + off); off += nx * 2;
  uint16_t* VtG  = (uint16_t*)(ws + off); off += nx * 2;
  uint16_t* Zb   = (uint16_t*)(ws + off); off += nx * 2;   // ~48 MB

  cvt_f32_bf16<<<(int)((nx + 255) / 256), 256, 0, stream>>>(x, xb, (int)nx);
  dim3 tb(32, 8), tg(D_ / 32, D_ / 32);
  cvt_t_f32_bf16<<<tg, tb, 0, stream>>>(Wq, Wqt, D_, D_);
  cvt_t_f32_bf16<<<tg, tb, 0, stream>>>(Wk, Wkt, D_, D_);
  cvt_t_f32_bf16<<<tg, tb, 0, stream>>>(Wv, Wvt, D_, D_);
  cvt_t_f32_bf16<<<tg, tb, 0, stream>>>(Wo, Wot, D_, D_);

  dim3 gg(D_ / 128, M / 128);            // 8 x 32
  gemm_bf16<0><<<gg, 256, 0, stream>>>(xb, Wqt, nullptr, Qb,  M, D_, D_);
  gemm_bf16<0><<<gg, 256, 0, stream>>>(xb, Wkt, nullptr, Kb,  M, D_, D_);
  gemm_bf16<1><<<gg, 256, 0, stream>>>(xb, Wvt, nullptr, VtG, M, D_, D_);

  dim3 ag(S_ / 64, H_, B_);              // 32 x 16 x 2
  attn_kernel<<<ag, 128, 0, stream>>>(Qb, Kb, VtG, Zb);

  gemm_bf16<2><<<gg, 256, 0, stream>>>(Zb, Wot, bo, out, M, D_, D_);
}
